// SparseTransformerEncoder_69020124446904
// MI455X (gfx1250) — matrix operand order repaired and checked by reading, not compiled, not run on hardware
//
#include <hip/hip_runtime.h>
#include <hip/hip_bf16.h>
#include <math.h>

// ---------------------------------------------------------------------------
// CDNA5 (gfx1250) sparse-transformer encoder forward.
// All GEMMs + attention matmuls use v_wmma_f32_16x16x32_bf16 (wave32).
// GEMM: 256x64 block tile, double-buffered LDS (one barrier / k-step),
// 32 rows per wave -> 8 back-to-back WMMAs per k-step per wave.
// All WMMA fragment loads are contiguous 2x16B (ds_load_b128 / global b128):
//  - weights pre-transposed to [N][K] during fp32->bf16 conversion
//  - V stored head-dim-major [B,H,HD,S] so the P*V B-fragment is contiguous
// ---------------------------------------------------------------------------

typedef __attribute__((ext_vector_type(16))) __bf16 v16bf;
typedef __attribute__((ext_vector_type(8)))  float  v8f;

#define B_    2
#define S_    2048
#define D_    512
#define H_    8
#define HD_   64
#define L_    4
#define MLP_  2048
#define BS_   128
#define C_    32
#define NB_   16
#define NTOK  (B_ * S_)          // 4096 rows
#define NEGBIG (-1.0e9f)

#define GBM 256
#define GBN 64
#define GBK 32

// ---- WMMA helpers ----------------------------------------------------------

__device__ __forceinline__ v8f wmma_bf16(v16bf a, v16bf b, v8f c) {
  // 8 args: (neg_a, A, neg_b, B, c_mod, C, reuse_a, reuse_b)
  return __builtin_amdgcn_wmma_f32_16x16x32_bf16(false, a, false, b,
                                                 (short)0, c, false, false);
}

// Two contiguous 16B loads -> one 16-element bf16 fragment register group.
__device__ __forceinline__ v16bf load16x2(const __bf16* p0, const __bf16* p1) {
  union { uint4 u[2]; v16bf v; } t;
  t.u[0] = *(const uint4*)p0;
  t.u[1] = *(const uint4*)p1;
  return t.v;
}

// A-matrix 16x32 bf16 fragment (same K order as the B fragments):
// lanes 0-15: row M=lane,   K in {0..15}
// lanes16-31: row M=lane-16, K in {16..31}
__device__ __forceinline__ v16bf load_a_frag(const __bf16* base, int ld, int lane) {
  const __bf16* p = base + (size_t)(lane & 15) * ld + ((lane >> 4) << 4);
  return load16x2(p, p + 8);
}

// B-matrix 32x16 bf16 fragment from K-contiguous (transposed, [N][K]) storage:
// lane holds column N = lane&15; lanes 0-15 K=0..15, lanes 16-31 K=16..31.
__device__ __forceinline__ v16bf load_bT_frag(const __bf16* base, int ldk, int lane) {
  const __bf16* p = base + (size_t)(lane & 15) * ldk + ((lane >> 4) << 4);
  return load16x2(p, p + 8);
}

// B fragment for K^T (scores = Q * K^T): column n = key, K-dim = head dim.
__device__ __forceinline__ v16bf load_kT_frag(const __bf16* kbp, int lane, int hdc) {
  const __bf16* p = kbp + (size_t)(lane & 15) * HD_ + hdc * 32 + ((lane >> 4) << 4);
  return load16x2(p, p + 8);
}

// B fragment for V (out = P * V) from head-dim-major vT[hd][s].
__device__ __forceinline__ v16bf load_vT_frag(const __bf16* vtb, int s0, int lane, int nt) {
  const __bf16* p = vtb + (size_t)(nt * 16 + (lane & 15)) * S_ + s0 + ((lane >> 4) << 4);
  return load16x2(p, p + 8);
}

__device__ __forceinline__ float gelu_f(float x) {
  const float x3 = x * x * x;
  return 0.5f * x * (1.0f + tanhf(0.7978845608028654f * (x + 0.044715f * x3)));
}

// ---- elementwise / prologue kernels ---------------------------------------

// fp32 [L][K][N] -> bf16 transposed [L][N][K]
__global__ __launch_bounds__(256) void cvt_t_bf16_kernel(const float* __restrict__ src,
                                                         __bf16* __restrict__ dst,
                                                         int K, int N, int total) {
  const int i = blockIdx.x * 256 + threadIdx.x;
  if (i >= total) return;
  const int kn = K * N;
  const int l = i / kn;
  const int r = i - l * kn;
  const int k = r / N;
  const int n = r - k * N;
  dst[(size_t)l * kn + (size_t)n * K + k] = (__bf16)src[i];
}

__global__ __launch_bounds__(256) void embed_pe_kernel(const int* __restrict__ inputs,
                                                       const float* __restrict__ embed,
                                                       float* __restrict__ x) {
  const int i = blockIdx.x * 256 + threadIdx.x;     // over NTOK*D
  const int t = i >> 9;                             // token row, D=512
  const int d = i & (D_ - 1);
  const int s = t & (S_ - 1);
  const int id = inputs[t];
  const float pos = (float)s;
  float pe;
  if (d < D_ / 2) {
    const float dv = __expf(-(logf(10000.0f) / (D_ / 2)) * (float)d);
    pe = __sinf(pos * dv);
  } else {
    const float dv = __expf(-(logf(10000.0f) / (D_ / 2)) * (float)(d - D_ / 2));
    pe = __cosf(pos * dv);
  }
  x[i] = embed[(size_t)id * D_ + d] + pe;
}

// LayerNorm over D=512; one row per block, 256 threads (2 elems/thread).
__global__ __launch_bounds__(256) void ln_kernel(const float* __restrict__ x,
                                                 const float* __restrict__ g,
                                                 const float* __restrict__ be,
                                                 float* __restrict__ outF,
                                                 __bf16* __restrict__ outB) {
  const int row = blockIdx.x;
  const int tid = threadIdx.x;
  const float* xr = x + (size_t)row * D_;
  const float v0 = xr[tid], v1 = xr[tid + 256];
  float s = v0 + v1, ss = v0 * v0 + v1 * v1;
  #pragma unroll
  for (int m = 1; m < 32; m <<= 1) {
    s  += __shfl_xor(s,  m, 32);
    ss += __shfl_xor(ss, m, 32);
  }
  __shared__ float rs[8], rss[8];
  if ((tid & 31) == 0) { rs[tid >> 5] = s; rss[tid >> 5] = ss; }
  __syncthreads();
  float S1 = 0.f, S2 = 0.f;
  #pragma unroll
  for (int i = 0; i < 8; ++i) { S1 += rs[i]; S2 += rss[i]; }
  const float mean = S1 * (1.0f / D_);
  const float var  = S2 * (1.0f / D_) - mean * mean;
  const float inv  = rsqrtf(var + 1e-6f);
  const float y0 = (v0 - mean) * inv * g[tid]       + be[tid];
  const float y1 = (v1 - mean) * inv * g[tid + 256] + be[tid + 256];
  if (outF) {
    outF[(size_t)row * D_ + tid]       = y0;
    outF[(size_t)row * D_ + tid + 256] = y1;
  }
  if (outB) {
    outB[(size_t)row * D_ + tid]       = (__bf16)y0;
    outB[(size_t)row * D_ + tid + 256] = (__bf16)y1;
  }
}

// ---- WMMA GEMM: out[M,N] = A[M,K](bf16) * Wt[N,K](bf16) [+bias][gelu][+resid]
#define GF_GELU    1
#define GF_QKV     2   // write bf16 to [B,H,S,HD] layout
#define GF_BF16OUT 4   // write bf16 row-major [M,N]
#define GF_RESID   8   // outF = resid + acc (f32)
#define GF_QKVT    16  // write bf16 to [B,H,HD,S] layout (head-dim-major, for V)

__global__ __launch_bounds__(256) void gemm_bf16_kernel(
    const __bf16* __restrict__ A, const __bf16* __restrict__ Wt,
    const float* __restrict__ bias, const float* __restrict__ resid,
    float* __restrict__ outF, __bf16* __restrict__ outB,
    int K, int N, int mode, float scale) {
  __shared__ __attribute__((aligned(16))) __bf16 sA[2][GBM][GBK];   // 2 x 16 KB
  __shared__ __attribute__((aligned(16))) __bf16 sBT[2][GBN][GBK];  // 2 x  4 KB
  const int tid  = threadIdx.x;
  const int lane = tid & 31, wid = tid >> 5;
  const int bm = blockIdx.x * GBM;
  const int bn = blockIdx.y * GBN;
  v8f acc[2][4] = {};

  // staging map: A tile 256x32 (16 KB) -> 4 x 16B chunks per thread
  //              B tile  64x32 ( 4 KB) -> 1 x 16B chunk per thread
  const int ar = tid >> 2;            // 0..63  (A rows ar, ar+64, ar+128, ar+192)
  const int ao = (tid & 3) * 8;       // element offset within row
  const int br = tid >> 2;            // B row 0..63
  const int bo = (tid & 3) * 8;

  const __bf16* aB = A  + (size_t)(bm + ar) * K + ao;
  const __bf16* bB = Wt + (size_t)(bn + br) * K + bo;

  // preload tile 0 into buffer 0
  {
    uint4 ra0 = *(const uint4*)(aB);
    uint4 ra1 = *(const uint4*)(aB + (size_t)64  * K);
    uint4 ra2 = *(const uint4*)(aB + (size_t)128 * K);
    uint4 ra3 = *(const uint4*)(aB + (size_t)192 * K);
    uint4 rb0 = *(const uint4*)(bB);
    *(uint4*)&sA[0][ar][ao]        = ra0;
    *(uint4*)&sA[0][ar + 64][ao]   = ra1;
    *(uint4*)&sA[0][ar + 128][ao]  = ra2;
    *(uint4*)&sA[0][ar + 192][ao]  = ra3;
    *(uint4*)&sBT[0][br][bo]       = rb0;
  }
  __syncthreads();

  const int iters = K / GBK;
  for (int it = 0; it < iters; ++it) {
    const int cur = it & 1, nxt = cur ^ 1;
    const bool more = (it + 1) < iters;
    uint4 ra0, ra1, ra2, ra3, rb0;
    if (more) {                       // issue next tile's global loads early
      const int kn = (it + 1) * GBK;
      ra0 = *(const uint4*)(aB + kn);
      ra1 = *(const uint4*)(aB + (size_t)64  * K + kn);
      ra2 = *(const uint4*)(aB + (size_t)128 * K + kn);
      ra3 = *(const uint4*)(aB + (size_t)192 * K + kn);
      rb0 = *(const uint4*)(bB + kn);
      if (it + 2 < iters) {           // global_prefetch_b8 for tile it+2
        __builtin_prefetch(aB + kn + GBK, 0, 1);
        __builtin_prefetch(bB + kn + GBK, 0, 1);
      }
    }
    // compute current tile: 2 A-frags x 4 B-frags -> 8 back-to-back WMMAs
    const __bf16* aw = &sA[cur][wid * 32][0];
    const v16bf af0 = load_a_frag(aw, GBK, lane);
    const v16bf af1 = load_a_frag(aw + 16 * GBK, GBK, lane);
    v16bf bfr[4];
    #pragma unroll
    for (int nt = 0; nt < 4; ++nt)
      bfr[nt] = load_bT_frag(&sBT[cur][nt * 16][0], GBK, lane);
    #pragma unroll
    for (int nt = 0; nt < 4; ++nt)
      acc[0][nt] = wmma_bf16(af0, bfr[nt], acc[0][nt]);
    #pragma unroll
    for (int nt = 0; nt < 4; ++nt)
      acc[1][nt] = wmma_bf16(af1, bfr[nt], acc[1][nt]);
    if (more) {                       // stage next tile into the other buffer
      *(uint4*)&sA[nxt][ar][ao]       = ra0;
      *(uint4*)&sA[nxt][ar + 64][ao]  = ra1;
      *(uint4*)&sA[nxt][ar + 128][ao] = ra2;
      *(uint4*)&sA[nxt][ar + 192][ao] = ra3;
      *(uint4*)&sBT[nxt][br][bo]      = rb0;
    }
    __syncthreads();
  }

  // C/D layout: VGPR j, lanes0-15 -> (M=j, N=lane); lanes16-31 -> (M=j+8, N=lane-16)
  const int rl = (lane >> 4) << 3;
  const int cl = lane & 15;
  #pragma unroll
  for (int mt = 0; mt < 2; ++mt) {
    #pragma unroll
    for (int nt = 0; nt < 4; ++nt) {
      const int col = bn + nt * 16 + cl;
      const float bv = bias ? bias[col] : 0.0f;
      #pragma unroll
      for (int j = 0; j < 8; ++j) {
        const int row = bm + wid * 32 + mt * 16 + rl + j;
        float val = acc[mt][nt][j] * scale + bv;
        if (mode & GF_GELU) val = gelu_f(val);
        if (mode & GF_QKV) {
          const int bb = row >> 11, ssx = row & (S_ - 1);
          const int hh = col >> 6,  hd  = col & (HD_ - 1);
          outB[(((size_t)bb * H_ + hh) * S_ + ssx) * HD_ + hd] = (__bf16)val;
        } else if (mode & GF_QKVT) {
          const int bb = row >> 11, ssx = row & (S_ - 1);
          const int hh = col >> 6,  hd  = col & (HD_ - 1);
          outB[(((size_t)bb * H_ + hh) * HD_ + hd) * S_ + ssx] = (__bf16)val;
        } else if (mode & GF_BF16OUT) {
          outB[(size_t)row * N + col] = (__bf16)val;
        } else {  // GF_RESID
          const size_t idx = (size_t)row * N + col;
          outF[idx] = resid[idx] + val;
        }
      }
    }
  }
}

// ---- Block-sparse attention (flash-style online softmax, WMMA) -------------
// Grid: B*H*NB blocks; 256 threads = 8 waves; wave w owns 16 query rows.
__global__ __launch_bounds__(256) void attn_kernel(
    const __bf16* __restrict__ q, const __bf16* __restrict__ k,
    const __bf16* __restrict__ vT, const int* __restrict__ inputs,
    __bf16* __restrict__ o) {
  const int blk = blockIdx.x;
  const int nb = blk & (NB_ - 1);
  const int h  = (blk >> 4) & (H_ - 1);
  const int b  = blk >> 7;
  const int tid = threadIdx.x;
  const int lane = tid & 31, wid = tid >> 5;
  const int rl = (lane >> 4) << 3;
  const int cl = lane & 15;

  __shared__ unsigned char padL[BS_];
  __shared__ unsigned char padS[NB_ * C_];
  __shared__ __attribute__((aligned(16))) __bf16 sP[8][16][32];   // 8 KB

  for (int i = tid; i < BS_; i += 256)
    padL[i] = (inputs[b * S_ + nb * BS_ + i] > 0) ? 1 : 0;
  for (int i = tid; i < NB_ * C_; i += 256) {
    const int jb = i >> 5;
    const int sg = jb * BS_ + (BS_ - C_) + (i & (C_ - 1));
    padS[i] = (inputs[b * S_ + sg] > 0) ? 1 : 0;
  }
  __syncthreads();

  const size_t head = ((size_t)(b * H_ + h)) * S_ * HD_;   // q/k [token][HD]
  const __bf16* vtb = vT + head;                           // vT  [HD][token]
  const int q0 = nb * BS_ + wid * 16;
  const __bf16* qrowp = q + head + (size_t)q0 * HD_;
  const v16bf aq0 = load_a_frag(qrowp, HD_, lane);        // K = hd 0..31
  const v16bf aq1 = load_a_frag(qrowp + 32, HD_, lane);   // K = hd 32..63

  v8f oc[4] = {};
  float mrun[8], lrun[8];
  #pragma unroll
  for (int j = 0; j < 8; ++j) { mrun[j] = -3.0e38f; lrun[j] = 0.0f; }

  // Key pairs: 4 local (128 keys) then nb summary pairs (32 contiguous tokens
  // each: last 32 tokens of block jb). Pairs with jb>=nb are fully masked ->
  // exp() == 0, so they are skipped.
  const int npairs = 4 + nb;
  for (int p = 0; p < npairs; ++p) {
    const bool local = (p < 4);
    const int jb = p - 4;
    const int s0 = local ? (nb * BS_ + p * 32) : (jb * BS_ + (BS_ - C_));
    const __bf16* kbase = k + head + (size_t)s0 * HD_;

    v8f sc0 = {}, sc1 = {};
    sc0 = wmma_bf16(aq0, load_kT_frag(kbase, lane, 0), sc0);
    sc0 = wmma_bf16(aq1, load_kT_frag(kbase, lane, 1), sc0);
    sc1 = wmma_bf16(aq0, load_kT_frag(kbase + 16 * HD_, lane, 0), sc1);
    sc1 = wmma_bf16(aq1, load_kT_frag(kbase + 16 * HD_, lane, 1), sc1);

    float alphaj[8];
    #pragma unroll
    for (int j = 0; j < 8; ++j) {
      const int qrow = wid * 16 + rl + j;       // query pos within block
      bool ok0, ok1;
      if (local) {
        const int kp0 = p * 32 + cl, kp1 = kp0 + 16;
        ok0 = (kp0 <= qrow) && padL[kp0];
        ok1 = (kp1 <= qrow) && padL[kp1];
      } else {
        const int m0 = jb * 32 + cl;
        ok0 = padS[m0] != 0;
        ok1 = padS[m0 + 16] != 0;
      }
      const float s0v = ok0 ? (float)sc0[j] : NEGBIG;
      const float s1v = ok1 ? (float)sc1[j] : NEGBIG;
      float tmax = fmaxf(s0v, s1v);
      #pragma unroll
      for (int m = 1; m < 16; m <<= 1) tmax = fmaxf(tmax, __shfl_xor(tmax, m, 32));
      const float mnew  = fmaxf(mrun[j], tmax);
      const float alpha = __expf(mrun[j] - mnew);
      const float p0 = __expf(s0v - mnew);
      const float p1 = __expf(s1v - mnew);
      float psum = p0 + p1;
      #pragma unroll
      for (int m = 1; m < 16; m <<= 1) psum += __shfl_xor(psum, m, 32);
      lrun[j] = lrun[j] * alpha + psum;
      mrun[j] = mnew;
      alphaj[j] = alpha;
      sP[wid][rl + j][cl]      = (__bf16)p0;    // transpose bounce for A-frag
      sP[wid][rl + j][cl + 16] = (__bf16)p1;
    }
    #pragma unroll
    for (int nt = 0; nt < 4; ++nt)
      #pragma unroll
      for (int j = 0; j < 8; ++j) oc[nt][j] *= alphaj[j];
    __syncthreads();
    const v16bf aP = load_a_frag(&sP[wid][0][0], 32, lane);
    #pragma unroll
    for (int nt = 0; nt < 4; ++nt)
      oc[nt] = wmma_bf16(aP, load_vT_frag(vtb, s0, lane, nt), oc[nt]);
    __syncthreads();
  }

  #pragma unroll
  for (int j = 0; j < 8; ++j) {
    const float inv = 1.0f / fmaxf(lrun[j], 1e-20f);
    const int srow = nb * BS_ + wid * 16 + rl + j;
    __bf16* op = o + ((size_t)(b * S_ + srow)) * D_ + h * HD_;
    #pragma unroll
    for (int nt = 0; nt < 4; ++nt)
      op[nt * 16 + cl] = (__bf16)(oc[nt][j] * inv);
  }
}

// ---------------------------------------------------------------------------

extern "C" void kernel_launch(void* const* d_in, const int* in_sizes, int n_in,
                              void* d_out, int out_size, void* d_ws, size_t ws_size,
                              hipStream_t stream) {
  (void)in_sizes; (void)n_in; (void)out_size; (void)ws_size;
  const int*   inputs = (const int*)  d_in[0];
  const float* embed  = (const float*)d_in[1];
  const float* ln1_s  = (const float*)d_in[2];
  const float* ln1_b  = (const float*)d_in[3];
  const float* wq     = (const float*)d_in[4];
  const float* wk     = (const float*)d_in[5];
  const float* wv     = (const float*)d_in[6];
  const float* wo     = (const float*)d_in[7];
  const float* ln2_s  = (const float*)d_in[8];
  const float* ln2_b  = (const float*)d_in[9];
  const float* w1     = (const float*)d_in[10];
  const float* b1     = (const float*)d_in[11];
  const float* w2     = (const float*)d_in[12];
  const float* b2     = (const float*)d_in[13];
  const float* lnf_s  = (const float*)d_in[14];
  const float* lnf_b  = (const float*)d_in[15];

  char* ws = (char*)d_ws;
  size_t off = 0;
  auto alloc = [&](size_t bytes) -> void* {
    void* p = ws + off;
    off += (bytes + 255) & ~(size_t)255;
    return p;
  };
  float*  x   = (float*) alloc((size_t)NTOK * D_ * 4);             // 8 MB
  __bf16* hb  = (__bf16*)alloc((size_t)NTOK * D_ * 2);             // 4 MB
  __bf16* qb  = (__bf16*)alloc((size_t)B_ * H_ * S_ * HD_ * 2);    // 4 MB
  __bf16* kb  = (__bf16*)alloc((size_t)B_ * H_ * S_ * HD_ * 2);
  __bf16* vtb = (__bf16*)alloc((size_t)B_ * H_ * HD_ * S_ * 2);    // head-dim-major
  __bf16* ob  = (__bf16*)alloc((size_t)NTOK * D_ * 2);             // 4 MB
  __bf16* m1b = (__bf16*)alloc((size_t)NTOK * MLP_ * 2);           // 16 MB
  __bf16* wqt = (__bf16*)alloc((size_t)L_ * D_ * D_ * 2);          // [N][K] layouts
  __bf16* wkt = (__bf16*)alloc((size_t)L_ * D_ * D_ * 2);
  __bf16* wvt = (__bf16*)alloc((size_t)L_ * D_ * D_ * 2);
  __bf16* wot = (__bf16*)alloc((size_t)L_ * D_ * D_ * 2);
  __bf16* w1t = (__bf16*)alloc((size_t)L_ * D_ * MLP_ * 2);        // 8 MB
  __bf16* w2t = (__bf16*)alloc((size_t)L_ * MLP_ * D_ * 2);        // 8 MB

  // --- weight conversion fp32 -> bf16 with [K][N] -> [N][K] transpose ---
  const int nWsq = L_ * D_ * D_;          // 1,048,576
  const int nWm  = L_ * D_ * MLP_;        // 4,194,304
  cvt_t_bf16_kernel<<<(nWsq + 255) / 256, 256, 0, stream>>>(wq, wqt, D_, D_, nWsq);
  cvt_t_bf16_kernel<<<(nWsq + 255) / 256, 256, 0, stream>>>(wk, wkt, D_, D_, nWsq);
  cvt_t_bf16_kernel<<<(nWsq + 255) / 256, 256, 0, stream>>>(wv, wvt, D_, D_, nWsq);
  cvt_t_bf16_kernel<<<(nWsq + 255) / 256, 256, 0, stream>>>(wo, wot, D_, D_, nWsq);
  cvt_t_bf16_kernel<<<(nWm  + 255) / 256, 256, 0, stream>>>(w1, w1t, D_, MLP_, nWm);
  cvt_t_bf16_kernel<<<(nWm  + 255) / 256, 256, 0, stream>>>(w2, w2t, MLP_, D_, nWm);

  // --- embedding + positional encoding ---
  embed_pe_kernel<<<(NTOK * D_) / 256, 256, 0, stream>>>(inputs, embed, x);

  const dim3 gemmD(NTOK / GBM, D_ / GBN);     // 16 x 8
  const dim3 gemmM(NTOK / GBM, MLP_ / GBN);   // 16 x 32

  for (int l = 0; l < L_; ++l) {
    const size_t wsq = (size_t)l * D_ * D_;
    const size_t wm  = (size_t)l * D_ * MLP_;
    // ln1 -> hb (bf16)
    ln_kernel<<<NTOK, 256, 0, stream>>>(x, ln1_s + l * D_, ln1_b + l * D_,
                                        nullptr, hb);
    // QKV projections (q pre-scaled by 1/sqrt(HD); V stored head-dim-major)
    gemm_bf16_kernel<<<gemmD, 256, 0, stream>>>(hb, wqt + wsq, nullptr, nullptr,
                                                nullptr, qb, D_, D_, GF_QKV, 0.125f);
    gemm_bf16_kernel<<<gemmD, 256, 0, stream>>>(hb, wkt + wsq, nullptr, nullptr,
                                                nullptr, kb, D_, D_, GF_QKV, 1.0f);
    gemm_bf16_kernel<<<gemmD, 256, 0, stream>>>(hb, wvt + wsq, nullptr, nullptr,
                                                nullptr, vtb, D_, D_, GF_QKVT, 1.0f);
    // block-sparse attention
    attn_kernel<<<B_ * H_ * NB_, 256, 0, stream>>>(qb, kb, vtb, inputs, ob);
    // output projection + residual into x
    gemm_bf16_kernel<<<gemmD, 256, 0, stream>>>(ob, wot + wsq, nullptr, x,
                                                x, nullptr, D_, D_, GF_RESID, 1.0f);
    // ln2 -> hb
    ln_kernel<<<NTOK, 256, 0, stream>>>(x, ln2_s + l * D_, ln2_b + l * D_,
                                        nullptr, hb);
    // MLP: gelu(hb @ w1 + b1) -> m1b ; m1b @ w2 + b2 + residual -> x
    gemm_bf16_kernel<<<gemmM, 256, 0, stream>>>(hb, w1t + wm, b1 + l * MLP_,
                                                nullptr, nullptr, m1b,
                                                D_, MLP_, GF_GELU | GF_BF16OUT, 1.0f);
    gemm_bf16_kernel<<<gemmD, 256, 0, stream>>>(m1b, w2t + wm, b2 + l * D_,
                                                x, x, nullptr,
                                                MLP_, D_, GF_RESID, 1.0f);
  }
  // final LN -> d_out (f32)
  ln_kernel<<<NTOK, 256, 0, stream>>>(x, lnf_s, lnf_b, (float*)d_out, nullptr);
}
